// SpatialNL_19765439496468
// MI455X (gfx1250) — matrix operand order repaired and checked by reading, not compiled, not run on hardware
//
#include <hip/hip_runtime.h>

// ---------------------------------------------------------------------------
// SpatialNL attention block for MI455X (gfx1250, wave32, WMMA bf16).
// B=4, C_IN=256, C_P=128, N=4096.
//
// Softmax over axis=1 => per-key normalizer: y[c,n] = sum_m exp(s[n,m]-M[m])/Z[m] * g[c,m]
// Pass 0: pack weights into bf16 WMMA fragments.
// Pass 1: projections t,p,g (WMMA), stored FRAGMENT-LINEAR (each lane's v16bf
//         operand chunk contiguous -> consumer loads are 2x b128).
// Pass 2: per-column stats M,Z (WMMA + online max/sum, shfl + LDS combine).
// Pass 3: fused exp(*1/Z) + second GEMM; per-wave LDS transpose of exp(s)
//         (scatter-write D-layout, vector-read A-layout, s_wait_dscnt only).
// Pass 4: z = Wz*y (WMMA) -> BN stats -> BN apply + residual.
// Working set ~34MB -> L2-resident (192MB); HBM ~= x in + out only.
// ---------------------------------------------------------------------------

typedef __attribute__((ext_vector_type(16))) __bf16 v16bf;
typedef __attribute__((ext_vector_type(8)))  float  v8f;

#define Bn  4
#define CIN 256
#define CP  128
#define NN  4096

__device__ inline v8f wmma_bf16(v16bf a, v16bf b, v8f c) {
  return __builtin_amdgcn_wmma_f32_16x16x32_bf16(false, a, false, b, (short)0, c,
                                                 false, false);
}

// A-matrix (bf16 16x32 MxK) lane layout chosen here (matches B's K order):
//   element e (0..15), half = lane>>4:  K = e + 16*half
__device__ inline int a_k_of(int e, int half) {
  return e + 16 * half;
}
// inverse: K -> (half, element)
__device__ inline void a_inv(int k, int& h, int& e) {
  h = k >> 4;
  e = k & 15;
}
// B-matrix (bf16 32x16 KxN) as packed here: lanes 0-15 K=0..15, lanes 16-31 K=16..31,
// col = lane&15, element e -> K = e + 16*half  (inverse: h=k>>4, e=k&15).
// C/D (f32 16x16): VGPR r -> row r + 8*half, col = lane&15.

// Fragment-linear indexing helpers (each fragment = 32 lanes * v16bf)
__device__ __host__ inline size_t fragA_t(int b, int nt, int ks) {   // tA / yA
  return ((size_t)(b * 256 + nt) * 4 + ks) * 32;
}
__device__ __host__ inline size_t fragB_p(int b, int mt, int ks) {   // pB
  return ((size_t)(b * 256 + mt) * 4 + ks) * 32;
}
__device__ __host__ inline size_t fragB_g(int b, int msi, int ct) {  // gB
  return ((size_t)(b * 128 + msi) * 8 + ct) * 32;
}

// ---------------------------------------------------------------------------
// Kernel 0: pack Wt/Wp/Wg (A-fragments, 8 ot x 8 ks) and Wz (B-fragments,
// 16 ot x 4 ks) into bf16 fragment-linear buffers.
// ---------------------------------------------------------------------------
__global__ __launch_bounds__(256) void prep_weights_kernel(
    const float* __restrict__ Wt, const float* __restrict__ Wp,
    const float* __restrict__ Wg, const float* __restrict__ Wz,
    __bf16* __restrict__ WtA, __bf16* __restrict__ WpA,
    __bf16* __restrict__ WgA, __bf16* __restrict__ WzB) {
  const int which = blockIdx.x;
  if (which < 3) {
    const float* W = (which == 0) ? Wt : (which == 1) ? Wp : Wg;
    __bf16* D = (which == 0) ? WtA : (which == 1) ? WpA : WgA;
    for (int i = threadIdx.x; i < 8 * 8 * 32 * 16; i += 256) {
      int e = i & 15, lane = (i >> 4) & 31, ks = (i >> 9) & 7, ot = i >> 12;
      int half = lane >> 4, row = ot * 16 + (lane & 15);
      int c = ks * 32 + a_k_of(e, half);
      D[i] = (__bf16)W[row * CIN + c];
    }
  } else {
    for (int i = threadIdx.x; i < 16 * 4 * 32 * 16; i += 256) {
      int e = i & 15, lane = (i >> 4) & 31, ks = (i >> 9) & 3, ot = i >> 11;
      int half = lane >> 4;
      int c = ks * 32 + e + 16 * half;      // K dim = input channel c
      int o = ot * 16 + (lane & 15);        // col = output channel o
      WzB[i] = (__bf16)Wz[o * CP + c];
    }
  }
}

// ---------------------------------------------------------------------------
// Kernel 1: t,p,g projections (K=256). x tile staged in LDS already in
// B-fragment layout; weight fragments are vector loads. Outputs scattered
// into fragment-linear tA / pB / gB.
// ---------------------------------------------------------------------------
__global__ __launch_bounds__(256) void proj_kernel(
    const float* __restrict__ x, const v16bf* __restrict__ WtA,
    const v16bf* __restrict__ WpA, const v16bf* __restrict__ WgA,
    __bf16* __restrict__ tA, __bf16* __restrict__ pB, __bf16* __restrict__ gB) {
  const int b = blockIdx.y;
  const int nbase = blockIdx.x * 16;
  const int nt = blockIdx.x;
  __shared__ __align__(32) v16bf xsv[8 * 32];   // x B-fragments, ks=0..7
  __bf16* xse = (__bf16*)xsv;
  const int tid = threadIdx.x;
  for (int i = tid; i < CIN * 16; i += 256) {
    int c = i >> 4, j = i & 15;
    float v = x[((size_t)b * CIN + c) * NN + nbase + j];
    int ks = c >> 5, k = c & 31, h = k >> 4, e = k & 15;
    xse[((ks * 32) + j + 16 * h) * 16 + e] = (__bf16)v;   // B layout scatter
  }
  __syncthreads();
  const int wave = tid >> 5, lane = tid & 31, half = lane >> 4, col = lane & 15;
  v8f at = {}, ap = {}, ag = {};
  for (int ks = 0; ks < 8; ks++) {
    v16bf fat = WtA[(wave * 8 + ks) * 32 + lane];
    v16bf fap = WpA[(wave * 8 + ks) * 32 + lane];
    v16bf fag = WgA[(wave * 8 + ks) * 32 + lane];
    v16bf fb  = xsv[ks * 32 + lane];
    at = wmma_bf16(fat, fb, at);
    ap = wmma_bf16(fap, fb, ap);
    ag = wmma_bf16(fag, fb, ag);
  }
  // scatter D fragments into consumer fragment-linear layouts
#pragma unroll
  for (int r = 0; r < 8; r++) {
    const int o = wave * 16 + r + 8 * half;   // channel (Cp)
    const int n = nbase + col;                // spatial
    // tA: A-frag (row=n&15, K=o&31) of tile (nt, o>>5)
    { int h2, e2; a_inv(o & 31, h2, e2);
      tA[(fragA_t(b, nt, o >> 5) + col + 16 * h2) * 16 + e2] = (__bf16)at[r]; }
    // pB: B-frag (K=o&31, col=n&15) of tile (nt, o>>5)
    { int k = o & 31, h2 = k >> 4, e2 = k & 15;
      pB[(fragB_p(b, nt, o >> 5) + col + 16 * h2) * 16 + e2] = (__bf16)ap[r]; }
    // gB: B-frag (K=n&31, col=o&15) of tile (n>>5, o>>4)
    { int k = n & 31, h2 = k >> 4, e2 = k & 15;
      gB[(fragB_g(b, n >> 5, o >> 4) + (o & 15) + 16 * h2) * 16 + e2] = (__bf16)ag[r]; }
  }
}

// ---------------------------------------------------------------------------
// Kernel 2: M[m] = max_n s[n,m], Z[m] = sum_n exp(s[n,m]-M[m]); s = t^T p.
// p fragments (fixed m tile) preloaded once; tA fragments are 2xb128 loads.
// ---------------------------------------------------------------------------
__global__ __launch_bounds__(256) void stats_kernel(
    const v16bf* __restrict__ tA, const v16bf* __restrict__ pB,
    float* __restrict__ Mb, float* __restrict__ Zb) {
  const int b = blockIdx.y, mt = blockIdx.x;
  const int tid = threadIdx.x, wave = tid >> 5, lane = tid & 31;
  const int col = lane & 15;
  v16bf fbs[4];
#pragma unroll
  for (int ks = 0; ks < 4; ks++) fbs[ks] = pB[fragB_p(b, mt, ks) + lane];
  float runM = -3.4e38f, runZ = 0.f;
  for (int nt = wave; nt < NN / 16; nt += 8) {
    if (nt + 8 < NN / 16)
      __builtin_prefetch(&tA[fragA_t(b, nt + 8, 0) + lane], 0, 1);
    v8f acc = {};
#pragma unroll
    for (int ks = 0; ks < 4; ks++)
      acc = wmma_bf16(tA[fragA_t(b, nt, ks) + lane], fbs[ks], acc);
    float tmax = acc[0];
#pragma unroll
    for (int r = 1; r < 8; r++) tmax = fmaxf(tmax, acc[r]);
    float newM = fmaxf(runM, tmax);
    float s = 0.f;
#pragma unroll
    for (int r = 0; r < 8; r++) s += __expf(acc[r] - newM);
    runZ = runZ * __expf(runM - newM) + s;
    runM = newM;
  }
  // lanes L and L^16 hold the same column (rows r / r+8): fuse
  float oM = __shfl_xor(runM, 16, 32);
  float oZ = __shfl_xor(runZ, 16, 32);
  float cM = fmaxf(runM, oM);
  float cZ = runZ * __expf(runM - cM) + oZ * __expf(oM - cM);
  __shared__ float sM[8][16], sZ[8][16];
  if (lane < 16) { sM[wave][col] = cM; sZ[wave][col] = cZ; }
  __syncthreads();
  if (tid < 16) {
    float M0 = sM[0][tid], Z0 = sZ[0][tid];
#pragma unroll
    for (int w = 1; w < 8; w++) {
      float nM = fmaxf(M0, sM[w][tid]);
      Z0 = Z0 * __expf(M0 - nM) + sZ[w][tid] * __expf(sM[w][tid] - nM);
      M0 = nM;
    }
    Mb[(size_t)b * NN + mt * 16 + tid] = M0;
    Zb[(size_t)b * NN + mt * 16 + tid] = Z0;
  }
}

// ---------------------------------------------------------------------------
// Kernel 3: yT[n][c] = sum_m (exp(s[n,m]-M[m])/Z[m]) * g[c,m]
// 1/Z folded into exp (per-lane scalar). No block barriers: p/g fragments are
// vector loads from L2; only a per-wave 1KB LDS slab transposes exp(s) from
// D-layout to A-layout (DS ops are wave-ordered; one s_wait_dscnt needed).
// ---------------------------------------------------------------------------
__global__ __launch_bounds__(256) void apply_kernel(
    const v16bf* __restrict__ tA, const v16bf* __restrict__ pB,
    const v16bf* __restrict__ gB, const float* __restrict__ Mb,
    const float* __restrict__ Zb, __bf16* __restrict__ yA) {
  const int b = blockIdx.y;
  const int tid = threadIdx.x, wave = tid >> 5, lane = tid & 31;
  const int half = lane >> 4, col = lane & 15;
  const int nt = blockIdx.x * 8 + wave;

  __shared__ __align__(32) v16bf esv[8 * 32];   // per-wave exp(s) A-fragments
  __bf16* ese = (__bf16*)esv;

  v16bf fat[4];
#pragma unroll
  for (int ks = 0; ks < 4; ks++) fat[ks] = tA[fragA_t(b, nt, ks) + lane];

  v8f yacc[8];
#pragma unroll
  for (int i = 0; i < 8; i++) yacc[i] = (v8f){};

  for (int msi = 0; msi < NN / 32; msi++) {
    if (msi + 1 < NN / 32) {
      __builtin_prefetch(&pB[fragB_p(b, (msi + 1) * 2, 0) + lane], 0, 1);
      __builtin_prefetch(&gB[fragB_g(b, msi + 1, 0) + lane], 0, 1);
    }
#pragma unroll
    for (int sub = 0; sub < 2; sub++) {
      const int mt = msi * 2 + sub;
      v8f sacc = {};
#pragma unroll
      for (int ks = 0; ks < 4; ks++)
        sacc = wmma_bf16(fat[ks], pB[fragB_p(b, mt, ks) + lane], sacc);
      const float Mcol = Mb[(size_t)b * NN + mt * 16 + col];
      const float rz   = 1.0f / Zb[(size_t)b * NN + mt * 16 + col];
      // scatter exp into A-layout: row = n&15 = r+8*half, K = sub*16+col
      const int k = sub * 16 + col;
      int h2, e2; a_inv(k, h2, e2);
#pragma unroll
      for (int r = 0; r < 8; r++) {
        float ev = __expf(sacc[r] - Mcol) * rz;
        ese[((wave * 32) + (r + 8 * half) + 16 * h2) * 16 + e2] = (__bf16)ev;
      }
    }
    asm volatile("s_wait_dscnt 0" ::: "memory");   // wave-local LDS RAW
    v16bf ea = esv[wave * 32 + lane];
#pragma unroll
    for (int ct = 0; ct < 8; ct++)
      yacc[ct] = wmma_bf16(ea, gB[fragB_g(b, msi, ct) + lane], yacc[ct]);
  }
  // store yT as A-fragments for the z GEMM: rows = n, K = c
#pragma unroll
  for (int ct = 0; ct < 8; ct++) {
#pragma unroll
    for (int r = 0; r < 8; r++) {
      const int c = ct * 16 + col;
      int h2, e2; a_inv(c & 31, h2, e2);
      yA[(fragA_t(b, nt, c >> 5) + (r + 8 * half) + 16 * h2) * 16 + e2] =
          (__bf16)yacc[ct][r];
    }
  }
}

// ---------------------------------------------------------------------------
// Kernel 4a: zT[n][o] = sum_c yT[n][c] * Wz[o][c]  (K=128, WMMA)
// ---------------------------------------------------------------------------
__global__ __launch_bounds__(256) void zgemm_kernel(
    const v16bf* __restrict__ yA, const v16bf* __restrict__ WzB,
    float* __restrict__ zT) {
  const int b = blockIdx.y, nt = blockIdx.x;
  const int tid = threadIdx.x, wave = tid >> 5, lane = tid & 31;
  const int half = lane >> 4, col = lane & 15;
  v16bf fay[4];
#pragma unroll
  for (int ks = 0; ks < 4; ks++) fay[ks] = yA[fragA_t(b, nt, ks) + lane];
#pragma unroll
  for (int sub = 0; sub < 2; sub++) {
    const int ot = wave * 2 + sub;            // 8 waves x 2 = 256 channels
    v8f acc = {};
#pragma unroll
    for (int ks = 0; ks < 4; ks++)
      acc = wmma_bf16(fay[ks], WzB[(ot * 4 + ks) * 32 + lane], acc);
#pragma unroll
    for (int r = 0; r < 8; r++)
      zT[((size_t)b * NN + nt * 16 + r + 8 * half) * CIN + ot * 16 + col] = acc[r];
  }
}

// ---------------------------------------------------------------------------
// Kernel 4b: BN batch stats per channel
// ---------------------------------------------------------------------------
__global__ __launch_bounds__(256) void bnstats_kernel(
    const float* __restrict__ zT, float* __restrict__ meanb,
    float* __restrict__ rstdb) {
  const int o = blockIdx.x;
  float s = 0.f, sq = 0.f;
  for (int i = threadIdx.x; i < Bn * NN; i += 256) {
    float v = zT[(size_t)i * CIN + o];
    s += v; sq += v * v;
  }
  __shared__ float ss[256], qq[256];
  ss[threadIdx.x] = s; qq[threadIdx.x] = sq;
  __syncthreads();
  for (int st = 128; st > 0; st >>= 1) {
    if (threadIdx.x < st) {
      ss[threadIdx.x] += ss[threadIdx.x + st];
      qq[threadIdx.x] += qq[threadIdx.x + st];
    }
    __syncthreads();
  }
  if (threadIdx.x == 0) {
    const float inv = 1.0f / (float)(Bn * NN);
    float mu  = ss[0] * inv;
    float var = qq[0] * inv - mu * mu;
    meanb[o] = mu;
    rstdb[o] = rsqrtf(var + 1e-5f);
  }
}

// ---------------------------------------------------------------------------
// Kernel 4c: out = (z - mean)*rstd*gamma + beta + x   ([B][CIN][N])
// ---------------------------------------------------------------------------
__global__ __launch_bounds__(256) void bnout_kernel(
    const float* __restrict__ zT, const float* __restrict__ x,
    const float* __restrict__ gamma, const float* __restrict__ beta,
    const float* __restrict__ meanb, const float* __restrict__ rstdb,
    float* __restrict__ out) {
  const int o = blockIdx.x, b = blockIdx.y;
  const float mu = meanb[o], rs = rstdb[o], ga = gamma[o], be = beta[o];
  for (int n = threadIdx.x; n < NN; n += 256) {
    float z = zT[((size_t)b * NN + n) * CIN + o];
    size_t xi = ((size_t)b * CIN + o) * NN + n;
    out[xi] = (z - mu) * rs * ga + be + x[xi];
  }
}

// ---------------------------------------------------------------------------
extern "C" void kernel_launch(void* const* d_in, const int* in_sizes, int n_in,
                              void* d_out, int out_size, void* d_ws, size_t ws_size,
                              hipStream_t stream) {
  const float* x     = (const float*)d_in[0];
  const float* Wt    = (const float*)d_in[1];
  const float* Wp    = (const float*)d_in[2];
  const float* Wg    = (const float*)d_in[3];
  const float* Wz    = (const float*)d_in[4];
  const float* gamma = (const float*)d_in[5];
  const float* beta  = (const float*)d_in[6];
  float* out = (float*)d_out;

  char* w = (char*)d_ws;
  size_t off = 0;
  auto take = [&](size_t bytes) {
    void* p = w + off;
    off += (bytes + 255) & ~(size_t)255;
    return p;
  };
  __bf16* tA  = (__bf16*)take((size_t)Bn * 256 * 4 * 32 * 32);   // 4MB
  __bf16* pB  = (__bf16*)take((size_t)Bn * 256 * 4 * 32 * 32);   // 4MB
  __bf16* gB  = (__bf16*)take((size_t)Bn * 128 * 8 * 32 * 32);   // 4MB
  __bf16* yA  = (__bf16*)take((size_t)Bn * 256 * 4 * 32 * 32);   // 4MB
  __bf16* WtA = (__bf16*)take((size_t)8 * 8 * 32 * 32);          // 64KB
  __bf16* WpA = (__bf16*)take((size_t)8 * 8 * 32 * 32);
  __bf16* WgA = (__bf16*)take((size_t)8 * 8 * 32 * 32);
  __bf16* WzB = (__bf16*)take((size_t)16 * 4 * 32 * 32);
  float*  Mb  = (float*)take((size_t)Bn * NN * 4);
  float*  Zb  = (float*)take((size_t)Bn * NN * 4);
  float*  mnb = (float*)take(CIN * 4);
  float*  rsb = (float*)take(CIN * 4);
  float*  zT  = (float*)take((size_t)Bn * NN * CIN * 4);         // 16MB
  (void)ws_size; (void)in_sizes; (void)n_in; (void)out_size;

  prep_weights_kernel<<<dim3(4), 256, 0, stream>>>(Wt, Wp, Wg, Wz, WtA, WpA, WgA, WzB);
  proj_kernel<<<dim3(NN / 16, Bn), 256, 0, stream>>>(
      x, (const v16bf*)WtA, (const v16bf*)WpA, (const v16bf*)WgA, tA, pB, gB);
  stats_kernel<<<dim3(NN / 16, Bn), 256, 0, stream>>>(
      (const v16bf*)tA, (const v16bf*)pB, Mb, Zb);
  apply_kernel<<<dim3(NN / 128, Bn), 256, 0, stream>>>(
      (const v16bf*)tA, (const v16bf*)pB, (const v16bf*)gB, Mb, Zb, yA);
  zgemm_kernel<<<dim3(NN / 16, Bn), 256, 0, stream>>>(
      (const v16bf*)yA, (const v16bf*)WzB, zT);
  bnstats_kernel<<<dim3(CIN), 256, 0, stream>>>(zT, mnb, rsb);
  bnout_kernel<<<dim3(CIN, Bn), 256, 0, stream>>>(zT, x, gamma, beta, mnb, rsb, out);
}
